// InteractionLayer_24017457119876
// MI455X (gfx1250) — matrix operand order repaired and checked by reading, not compiled, not run on hardware
//
#include <hip/hip_runtime.h>
#include <hip/hip_bf16.h>

typedef __attribute__((ext_vector_type(16))) _Float16 v16h;
typedef __attribute__((ext_vector_type(8)))  _Float16 v8h;
typedef __attribute__((ext_vector_type(4)))  _Float16 v4h;
typedef __attribute__((ext_vector_type(8)))  float    v8f;
typedef __attribute__((ext_vector_type(4)))  float    v4f;

static constexpr int   Nn  = 8192;   // nodes
static constexpr int   Dd  = 128;    // D_IN == D_OUT
static constexpr float CUT = 0.5f;

union Frag16 { v16h v; v8h h[2]; };

// ---------------------------------------------------------------------------
// Kernel A: h = z @ W + B (fp32 accumulate), stored TRANSPOSED as f16:
//   hT[d][j] = (f16)( sum_k z[j][k] * W[k][d] + B[d] )
// ---------------------------------------------------------------------------
__global__ void __launch_bounds__(Dd)
build_hT_kernel(const float* __restrict__ z, const float* __restrict__ W,
                const float* __restrict__ Bv, _Float16* __restrict__ hT) {
    __shared__ float zrow[Dd];
    const int j = blockIdx.x;
    const int d = threadIdx.x;
    zrow[d] = z[(size_t)j * Dd + d];
    __syncthreads();
    float acc = Bv[d];
    #pragma unroll 8
    for (int k = 0; k < Dd; ++k)
        acc = fmaf(zrow[k], W[(size_t)k * Dd + d], acc);
    hT[(size_t)d * Nn + j] = (_Float16)acc;
}

// ---------------------------------------------------------------------------
// Kernel B: fused  out = S(dist) @ h,  K-tile = 64, software-pipelined with
// a 2-stage register ping-pong (no copy movs, no WMMA->VALU WAR nops).
// Block = 256 threads (8 waves); block owns 16 output rows, wave w owns
// output cols [16w,16w+16).  Per K-tile: each thread loads one float4 of
// dist (b128), computes 4 branchless masked sensitivities (v_rcp + v_exp +
// v_cndmask), packs to f16 LDS; each wave issues 2 WMMAs.  Next tile's dist
// and B fragments load during the current tile's barrier+WMMA window.
// ---------------------------------------------------------------------------
__global__ void __launch_bounds__(256)
interaction_kernel(const float* __restrict__ dist, const _Float16* __restrict__ hT,
                   const float* __restrict__ mu_p, const float* __restrict__ sg_p,
                   float* __restrict__ out) {
    constexpr int KT = 64;                         // K-tile
    constexpr int P  = 72;                         // LDS pitch (halves) = 36 DW:
                                                   // b128 A-loads bank-conflict-free
    __shared__ _Float16 sA[16 * P];

    const int tid  = threadIdx.x;
    const int lane = tid & 31;
    const int wave = tid >> 5;
    const int i0   = blockIdx.x * 16;              // output row tile
    const int n0   = wave * 16;                    // output col tile

    const float mu     = mu_p[0];
    const float sg     = sg_p[0];
    const float inv_mu = __builtin_amdgcn_rcpf(mu);
    // fold log2(e) into 1/(2*sigma^2): S = exp2(-(t^2) * c2)
    const float c2     = __builtin_amdgcn_rcpf(2.0f * sg * sg) * 1.4426950408889634f;

    // S-tile mapping: 16 rows x 64 cols, one float4 per thread
    const int srow = tid >> 4;                     // 0..15
    const int scol = (tid & 15) * 4;               // 0,4,...,60
    const int gi   = i0 + srow;
    const float* dRow = dist + (size_t)gi * Nn + scol;

    // A operand as loaded here (f16 16x32, k order matches B): lane -> row M=lane%16,
    // 16 contiguous halves starting at K = lane<16 ? 0 : 16
    const int aOff = (lane & 15) * P + ((lane < 16) ? 0 : 16);
    // B operand as loaded here (f16 32x16, k order matches A): lane -> col N=lane%16,
    // 16 contiguous halves starting at K = lane<16 ? 0 : 16
    const _Float16* bRow =
        hT + (size_t)(n0 + (lane & 15)) * Nn + ((lane < 16) ? 0 : 16);

    v8f acc = {};

    // one pipeline stage: consume (dc, bl0, bl1) for tile j0,
    // load tile jn into (dn, nb0, nb1)
    auto stage = [&](int j0, const v4f& dc, const Frag16& bl0, const Frag16& bl1,
                     int jn, v4f& dn, Frag16& nb0, Frag16& nb1)
                     __attribute__((always_inline)) {
        __syncthreads();                           // all waves done reading sA

        // ---- branchless S: v_rcp + v_exp + v_cndmask, no EXEC branching ----
        v4h sp;
        #pragma unroll
        for (int e = 0; e < 4; ++e) {
            const float d  = dc[e];
            const float t  = __builtin_amdgcn_rcpf(d) - inv_mu;
            const float ex = __builtin_amdgcn_exp2f(-(t * t) * c2);
            const bool  m  = (d < CUT) & ((j0 + scol + e) != gi);
            sp[e] = (_Float16)(m ? ex : 0.0f);
        }
        *(v4h*)&sA[srow * P + scol] = sp;

        // ---- issue next tile's global loads (hidden by barrier + WMMA) ----
        __builtin_prefetch(dRow + jn + KT, 0, 3);  // near-cache prefetch, 1 tile ahead
        dn = *(const v4f*)(dRow + jn);
        nb0.h[0] = *(const v8h*)(bRow + jn);      nb0.h[1] = *(const v8h*)(bRow + jn + 8);
        nb1.h[0] = *(const v8h*)(bRow + jn + 32); nb1.h[1] = *(const v8h*)(bRow + jn + 40);

        __syncthreads();                           // sA visible to all waves

        // ---- A fragments from LDS (conflict-free b128 loads) + 2 WMMAs ----
        Frag16 a0, a1;
        a0.h[0] = *(const v8h*)&sA[aOff];
        a0.h[1] = *(const v8h*)&sA[aOff + 8];
        a1.h[0] = *(const v8h*)&sA[aOff + 32];
        a1.h[1] = *(const v8h*)&sA[aOff + 40];

        acc = __builtin_amdgcn_wmma_f32_16x16x32_f16(
            false, a0.v, false, bl0.v, (short)0, acc, false, false);
        acc = __builtin_amdgcn_wmma_f32_16x16x32_f16(
            false, a1.v, false, bl1.v, (short)0, acc, false, false);
    };

    // ---- prologue: load tile 0 into register set A ----
    v4f dA, dB;
    Frag16 bA0, bA1, bB0, bB1;
    dA = *(const v4f*)dRow;
    bA0.h[0] = *(const v8h*)(bRow);      bA0.h[1] = *(const v8h*)(bRow + 8);
    bA1.h[0] = *(const v8h*)(bRow + 32); bA1.h[1] = *(const v8h*)(bRow + 40);

    // ---- main loop: 2 tiles per trip, ping-pong register sets ----
    for (int j0 = 0; j0 < Nn; j0 += 2 * KT) {
        const int j1 = j0 + KT;                    // Nn % 128 == 0 -> always valid
        const int j2 = (j1 + KT < Nn) ? (j1 + KT) : 0;  // clamp: last reload is dead
        stage(j0, dA, bA0, bA1, j1, dB, bB0, bB1);
        stage(j1, dB, bB0, bB1, j2, dA, bA0, bA1);
    }

    // C/D layout: VGPR r -> M = r + (lane<16 ? 0 : 8), N = lane%16
    const int cN = lane & 15;
    const int cM = (lane < 16) ? 0 : 8;
    #pragma unroll
    for (int r = 0; r < 8; ++r)
        out[(size_t)(i0 + cM + r) * Dd + n0 + cN] = acc[r];
}

// ---------------------------------------------------------------------------
// Host-side launcher
// inputs: 0:z [N,128] f32, 1:dist [N,N] f32, 2:W [128,128] f32, 3:B [128] f32,
//         4:mu [1] f32, 5:sigma [1] f32    output: [N,128] f32
// workspace: hT = f16 [128][8192] = 2 MB
// ---------------------------------------------------------------------------
extern "C" void kernel_launch(void* const* d_in, const int* in_sizes, int n_in,
                              void* d_out, int out_size, void* d_ws, size_t ws_size,
                              hipStream_t stream) {
    const float* z    = (const float*)d_in[0];
    const float* dist = (const float*)d_in[1];
    const float* W    = (const float*)d_in[2];
    const float* Bv   = (const float*)d_in[3];
    const float* mu   = (const float*)d_in[4];
    const float* sg   = (const float*)d_in[5];
    float*       out  = (float*)d_out;
    _Float16*    hT   = (_Float16*)d_ws;          // 128*8192*2 = 2 MB

    build_hT_kernel<<<Nn, Dd, 0, stream>>>(z, W, Bv, hT);
    interaction_kernel<<<Nn / 16, 256, 0, stream>>>(dist, hT, mu, sg, out);
}
